// MyRNN_68212670595860
// MI455X (gfx1250) — compile-verified, matrix operand order checked by reading, not run on hardware
//
#include <hip/hip_runtime.h>
#include <hip/hip_bf16.h>
#include <math.h>

typedef __attribute__((ext_vector_type(16))) __bf16 v16bf;
typedef __attribute__((ext_vector_type(8)))  float  v8f;

#define B_ 512
#define T_ 512
#define E_ 256
#define U_ 512
#define VOCAB_ 50000

#define NBLK 64
#define NTHR 512                    // 16 waves per block
#define NT_TILES 32                 // U_/16 tiles along N
#define KT1 24                      // (E_+U_)/32 k-tiles for layer 1
#define KT2 32                      // (U_+U_)/32 k-tiles for layer 2
// Block owns 4 batch-tiles x 4 N-tiles (64x64 output patch); 8x8 block grid.
#define LDS_DW ((KT1 * 4 + KT2 * 4) * 256)   // 57344 dwords = 224 KiB of weight tiles

#if __has_builtin(__builtin_amdgcn_tanhf)
#define TANH(x) __builtin_amdgcn_tanhf(x)
#elif __has_builtin(__builtin_amdgcn_tanh_f32)
#define TANH(x) __builtin_amdgcn_tanh_f32(x)
#else
#define TANH(x) tanhf(x)
#endif

static __device__ __forceinline__ unsigned f2bf1(float f) {
  unsigned u = __float_as_uint(f);
  return (u + 0x7FFFu + ((u >> 16) & 1u)) >> 16;  // RNE
}
static __device__ __forceinline__ unsigned packbf(float lo, float hi) {
  return f2bf1(lo) | (f2bf1(hi) << 16);
}

// A-fragment (16x32 bf16, wave32) from a bf16 row pointer.
// lane l: half=l>>4. v0..v3 <- K = kbase+half*8 .. +7 (16B contiguous)
//                    v4..v7 <- K = kbase+16+half*8 .. +7
static __device__ __forceinline__ v16bf loadA_row(const unsigned short* rowp,
                                                  int kbase, int half) {
  union { int4 q[2]; v16bf b; } u;
  const char* p = (const char*)(rowp + kbase + half * 8);
  u.q[0] = *(const int4*)p;
  u.q[1] = *(const int4*)(p + 32);  // +16 bf16 elements
  return u.b;
}

// B-fragment from LDS-resident packed tile (32B per lane, contiguous -> ds_load_b128).
static __device__ __forceinline__ v16bf loadB_lds(const unsigned* tp) {
  union { int4 q[2]; v16bf b; } u;
  const int4* q = (const int4*)tp;
  u.q[0] = q[0];
  u.q[1] = q[1];
  return u.b;
}

// One-time fp32 -> packed bf16 conversion of the embedding table (grid-stride).
__global__ void convert_emb(const float* __restrict__ src, unsigned* __restrict__ dst,
                            int npairs) {
  int i = blockIdx.x * blockDim.x + threadIdx.x;
  int stride = gridDim.x * blockDim.x;
  for (; i < npairs; i += stride) {
    float2 f = *(const float2*)(src + 2 * (size_t)i);
    dst[i] = packbf(f.x, f.y);
  }
}

// Pack [W1;U1] and [W2;U2] (row-major fp32 [K,N]) into WMMA B-fragment tiles (bf16).
// B-fragment (32x16): lane l: n=l&15, half=l>>4; dword j holds K pair
// (half*8+2j, +1) for j<4, (16+half*8+2(j-4), +1) for j>=4, column n.
__global__ void pack_weights(const float* __restrict__ W1, const float* __restrict__ U1,
                             const float* __restrict__ W2, const float* __restrict__ U2,
                             unsigned* __restrict__ Wp1, unsigned* __restrict__ Wp2) {
  int tid = blockIdx.x * blockDim.x + threadIdx.x;
  const int T1 = KT1 * NT_TILES, T2 = KT2 * NT_TILES;
  if (tid >= (T1 + T2) * 32) return;
  int lane = tid & 31;
  int tile = tid >> 5;
  int isW2 = (tile >= T1);
  int ltile = isW2 ? (tile - T1) : tile;
  unsigned* dst = (isW2 ? Wp2 : Wp1) + ((size_t)ltile * 32 + lane) * 8;
  int tn = ltile & 31;
  int kt = ltile >> 5;
  int n = tn * 16 + (lane & 15);
  int half = lane >> 4;
#pragma unroll
  for (int j = 0; j < 8; ++j) {
    int kloc = (j < 4) ? (half * 8 + 2 * j) : (16 + half * 8 + 2 * (j - 4));
    int k = kt * 32 + kloc;
    float lo, hi;
    if (!isW2) {  // rows 0..255 = W1, 256..767 = U1
      lo = (k     < E_) ? W1[(size_t)k * U_ + n]       : U1[(size_t)(k - E_) * U_ + n];
      hi = (k + 1 < E_) ? W1[(size_t)(k + 1) * U_ + n] : U1[(size_t)(k + 1 - E_) * U_ + n];
    } else {      // rows 0..511 = W2, 512..1023 = U2
      lo = (k     < U_) ? W2[(size_t)k * U_ + n]       : U2[(size_t)(k - U_) * U_ + n];
      hi = (k + 1 < U_) ? W2[(size_t)(k + 1) * U_ + n] : U2[(size_t)(k + 1 - U_) * U_ + n];
    }
    dst[j] = packbf(lo, hi);
  }
}

__global__ void init_state(unsigned* H0a, unsigned* H1a, unsigned* bar, int ndw) {
  int i = blockIdx.x * blockDim.x + threadIdx.x;
  if (i < ndw) { H0a[i] = 0u; H1a[i] = 0u; }
  if (i < 2) bar[i] = 0u;
}

static __device__ __forceinline__ void grid_barrier(unsigned* cnt, volatile unsigned* gen,
                                                    unsigned nblk) {
  __threadfence();          // release: make our stores visible device-wide
  __syncthreads();
  if (threadIdx.x == 0) {
    unsigned g = *gen;      // stable: gen only advances after ALL blocks add
    unsigned old = atomicAdd(cnt, 1u);
    if (old == nblk - 1u) {
      *cnt = 0u;
      __threadfence();
      atomicAdd((unsigned*)gen, 1u);
    } else {
      while (*gen == g) { __builtin_amdgcn_s_sleep(1); }
    }
  }
  __syncthreads();
  __threadfence();          // acquire
}

__global__ void __launch_bounds__(NTHR)
rnn_persistent(const int* __restrict__ tokens, const unsigned short* __restrict__ Ebf,
               const float* __restrict__ b1, const float* __restrict__ b2,
               const float* __restrict__ Wo, const float* __restrict__ bo,
               const unsigned* __restrict__ Wp1, const unsigned* __restrict__ Wp2,
               unsigned short* H0a, unsigned short* H0b,
               unsigned short* H1a, unsigned short* H1b,
               float* H1f, unsigned* bar_cnt, unsigned* bar_gen,
               float* __restrict__ out) {
  extern __shared__ unsigned smem[];  // 224 KiB: block's weight tiles, LDS-resident
  const int lane = threadIdx.x & 31;
  const int wv = threadIdx.x >> 5;    // 0..15
  const int btb = blockIdx.x >> 3;    // block's batch-patch 0..7
  const int btn = blockIdx.x & 7;     // block's N-patch     0..7
  const int wtb = wv >> 2;            // wave's batch tile within block 0..3
  const int wtn = wv & 3;             // wave's N tile within block     0..3
  const int tb = btb * 4 + wtb;       // global batch tile 0..31
  const int tn = btn * 4 + wtn;       // global N tile     0..31
  const int l15 = lane & 15;
  const int half = lane >> 4;
  const int b0 = tb * 16;
  const int n0 = tn * 16;
  const int arow = b0 + l15;          // A-fragment row (batch index) for this lane
  const int col = n0 + l15;           // C-fragment column for this lane
  const float bv1 = b1[col];
  const float bv2 = b2[col];

  // ---- Stage this block's weight tiles into LDS (once; reused for all 512 steps) ----
  // LDS layout: tiles [0, KT1*4)  = layer-1 (kt, wtn); [KT1*4, +KT2*4) = layer-2.
  for (int ti = wv; ti < KT1 * 4; ti += 16) {
    int kt = ti >> 2, tnl = ti & 3;
    const int4* src = (const int4*)(Wp1 + (((size_t)kt * NT_TILES + btn * 4 + tnl) * 32 + lane) * 8);
    int4* dst = (int4*)(smem + (size_t)ti * 256 + lane * 8);
    dst[0] = src[0];
    dst[1] = src[1];
  }
  for (int ti = wv; ti < KT2 * 4; ti += 16) {
    int kt = ti >> 2, tnl = ti & 3;
    const int4* src = (const int4*)(Wp2 + (((size_t)kt * NT_TILES + btn * 4 + tnl) * 32 + lane) * 8);
    int4* dst = (int4*)(smem + (size_t)(KT1 * 4) * 256 + (size_t)ti * 256 + lane * 8);
    dst[0] = src[0];
    dst[1] = src[1];
  }
  __syncthreads();
  const unsigned* wb1 = smem;                          // + (kt*4 + wtn)*256 + lane*8
  const unsigned* wb2 = smem + (size_t)(KT1 * 4) * 256;

  for (int t = 0; t < T_; ++t) {
    const unsigned short* H0r = (t & 1) ? H0b : H0a;
    unsigned short*       H0w = (t & 1) ? H0a : H0b;
    const unsigned short* H1r = (t & 1) ? H1b : H1a;
    unsigned short*       H1w = (t & 1) ? H1a : H1b;

    // ---- Phase 1: H0w = tanh( [x_t, H0r] @ [W1;U1] + b1 ) ----
    v8f acc = {0.f, 0.f, 0.f, 0.f, 0.f, 0.f, 0.f, 0.f};
    const int tok = tokens[arow * T_ + t];
    const unsigned short* erow = Ebf + (size_t)tok * E_;
    if (t + 1 < T_) {  // hide next step's gather latency behind this step's GEMMs
      const int tokn = tokens[arow * T_ + t + 1];
      __builtin_prefetch(Ebf + (size_t)tokn * E_, 0, 0);
    }
#pragma unroll
    for (int kt = 0; kt < 8; ++kt) {            // embedding part, K = 0..255
      v16bf a = loadA_row(erow, kt * 32, half);
      v16bf w = loadB_lds(wb1 + ((size_t)(kt * 4 + wtn)) * 256 + lane * 8);
      acc = __builtin_amdgcn_wmma_f32_16x16x32_bf16(false, a, false, w,
                                                    (short)0, acc, false, false);
    }
    const unsigned short* h0row = H0r + (size_t)arow * U_;
#pragma unroll 4
    for (int kt = 0; kt < 16; ++kt) {           // recurrent part, K = 256..767
      v16bf a = loadA_row(h0row, kt * 32, half);
      v16bf w = loadB_lds(wb1 + ((size_t)((8 + kt) * 4 + wtn)) * 256 + lane * 8);
      acc = __builtin_amdgcn_wmma_f32_16x16x32_bf16(false, a, false, w,
                                                    (short)0, acc, false, false);
    }
#pragma unroll
    for (int r = 0; r < 8; ++r) {
      float h = TANH(acc[r] + bv1);
      int mrow = b0 + r + 8 * half;
      H0w[(size_t)mrow * U_ + col] = (unsigned short)f2bf1(h);
    }

    grid_barrier(bar_cnt, bar_gen, NBLK);

    // ---- Phase 2: H1w = tanh( [H0w, H1r] @ [W2;U2] + b2 ) ----
    acc = (v8f){0.f, 0.f, 0.f, 0.f, 0.f, 0.f, 0.f, 0.f};
    const unsigned short* h0wrow = H0w + (size_t)arow * U_;
#pragma unroll 4
    for (int kt = 0; kt < 16; ++kt) {           // K = 0..511 from new H0
      v16bf a = loadA_row(h0wrow, kt * 32, half);
      v16bf w = loadB_lds(wb2 + ((size_t)(kt * 4 + wtn)) * 256 + lane * 8);
      acc = __builtin_amdgcn_wmma_f32_16x16x32_bf16(false, a, false, w,
                                                    (short)0, acc, false, false);
    }
    const unsigned short* h1row = H1r + (size_t)arow * U_;
#pragma unroll 4
    for (int kt = 0; kt < 16; ++kt) {           // K = 512..1023 from old H1
      v16bf a = loadA_row(h1row, kt * 32, half);
      v16bf w = loadB_lds(wb2 + ((size_t)((16 + kt) * 4 + wtn)) * 256 + lane * 8);
      acc = __builtin_amdgcn_wmma_f32_16x16x32_bf16(false, a, false, w,
                                                    (short)0, acc, false, false);
    }
#pragma unroll
    for (int r = 0; r < 8; ++r) {
      float h = TANH(acc[r] + bv2);
      int mrow = b0 + r + 8 * half;
      H1w[(size_t)mrow * U_ + col] = (unsigned short)f2bf1(h);
      if (t == T_ - 1) H1f[(size_t)mrow * U_ + col] = h;  // fp32 copy for output head
    }

    grid_barrier(bar_cnt, bar_gen, NBLK);
  }

  // ---- Output head: out[b] = sigmoid(H1_final[b,:] @ Wo + bo) ----
  if (blockIdx.x == 0) {
    int b = threadIdx.x;  // NTHR == B_ == 512
    float s = 0.f;
    for (int u = 0; u < U_; ++u) s += H1f[(size_t)b * U_ + u] * Wo[u];
    out[b] = 1.f / (1.f + expf(-(s + bo[0])));
  }
}

extern "C" void kernel_launch(void* const* d_in, const int* in_sizes, int n_in,
                              void* d_out, int out_size, void* d_ws, size_t ws_size,
                              hipStream_t stream) {
  const int*   tokens = (const int*)d_in[0];
  const float* emb    = (const float*)d_in[1];
  const float* W1     = (const float*)d_in[2];
  const float* U1     = (const float*)d_in[3];
  const float* b1     = (const float*)d_in[4];
  const float* W2     = (const float*)d_in[5];
  const float* U2     = (const float*)d_in[6];
  const float* b2     = (const float*)d_in[7];
  const float* Wo     = (const float*)d_in[8];
  const float* bo     = (const float*)d_in[9];
  float* out = (float*)d_out;

  char* ws = (char*)d_ws;
  size_t o = 0;
  unsigned* bar = (unsigned*)(ws + o); o += 256;
  unsigned* Wp1 = (unsigned*)(ws + o); o += (size_t)KT1 * NT_TILES * 32 * 8 * 4;  // 768 KiB
  unsigned* Wp2 = (unsigned*)(ws + o); o += (size_t)KT2 * NT_TILES * 32 * 8 * 4;  // 1 MiB
  unsigned short* H0a = (unsigned short*)(ws + o); o += (size_t)B_ * U_ * 2;
  unsigned short* H0b = (unsigned short*)(ws + o); o += (size_t)B_ * U_ * 2;
  unsigned short* H1a = (unsigned short*)(ws + o); o += (size_t)B_ * U_ * 2;
  unsigned short* H1b = (unsigned short*)(ws + o); o += (size_t)B_ * U_ * 2;
  float* H1f = (float*)(ws + o); o += (size_t)B_ * U_ * 4;
  unsigned short* Ebf = (unsigned short*)(ws + o); o += (size_t)VOCAB_ * E_ * 2;  // 25.6 MB
  (void)ws_size; (void)n_in; (void)in_sizes; (void)out_size;

  // One-time table conversions (off the critical path).
  const int npairs = VOCAB_ * E_ / 2;  // 6.4M dword stores
  convert_emb<<<4096, 256, 0, stream>>>(emb, (unsigned*)Ebf, npairs);

  const int packThreads = (KT1 * NT_TILES + KT2 * NT_TILES) * 32;  // 57344
  pack_weights<<<(packThreads + 255) / 256, 256, 0, stream>>>(W1, U1, W2, U2, Wp1, Wp2);

  const int ndw = B_ * U_ * 2 / 4;  // dwords per bf16 state buffer
  init_state<<<(ndw + 255) / 256, 256, 0, stream>>>((unsigned*)H0a, (unsigned*)H1a, bar, ndw);

  rnn_persistent<<<NBLK, NTHR, LDS_DW * 4, stream>>>(tokens, Ebf, b1, b2, Wo, bo, Wp1, Wp2,
                                                     H0a, H0b, H1a, H1b, H1f,
                                                     bar, bar + 1, out);
}
